// ContextMem_29059748725389
// MI455X (gfx1250) — matrix operand order repaired and checked by reading, not compiled, not run on hardware
//
#include <hip/hip_runtime.h>
#include <hip/hip_bf16.h>
#include <math.h>

#define B_  2
#define T_  2048
#define C_  1024
#define NH_ 16
#define HS_ 64
#define KEY_SCALE_MAX 11.090339660644531f   // log(2^16 - 1)

typedef __attribute__((ext_vector_type(16))) __bf16 v16bf;
typedef __attribute__((ext_vector_type(8)))  __bf16 v8bf;
typedef __attribute__((ext_vector_type(8)))  float  v8f;

union U16bf { v16bf v; v8bf h[2]; };

// A-fragment (16x32 bf16, MxK): lane L holds row M=L%16; elements 0..15 = K[base..base+16),
// base = (L/16)*16. Caller passes base-adjusted ptr.
static __device__ __forceinline__ v16bf load_a_frag(const __bf16* p) {
  U16bf u;
  u.h[0] = *(const v8bf*)(p);
  u.h[1] = *(const v8bf*)(p + 8);
  return u.v;
}

// ---------------------------------------------------------------------------
// fp32 -> bf16 elementwise convert
// ---------------------------------------------------------------------------
__global__ __launch_bounds__(256)
void f32_to_bf16_kernel(const float* __restrict__ src, __bf16* __restrict__ dst, int n) {
  int i = blockIdx.x * 256 + threadIdx.x;
  if (i < n) dst[i] = (__bf16)src[i];
}

// ---------------------------------------------------------------------------
// GEMM: out[m,n] = sum_k A[m,k] * W[n,k]
// A: (M,1024) bf16 row-major, W: (1024,1024) bf16 row-major, out: (M,1024) fp32.
// N = K = 1024 are compile-time constants so all strides fold into immediates.
// block = 256 threads = 8 waves; block tile 64(M) x 256(N); wave tile 32x64
// -> 8 v_wmma per 32-wide K-step (2 A-frags, 4 B-frags).
// ---------------------------------------------------------------------------
__global__ __launch_bounds__(256)
void gemm_bf16_nt_kernel(const __bf16* __restrict__ A, const __bf16* __restrict__ W,
                         float* __restrict__ out) {
  constexpr int K = 1024, N = 1024;
  const int wave = threadIdx.x >> 5;
  const int lane = threadIdx.x & 31;
  const int half = lane >> 4;
  const int l16  = lane & 15;
  const int m0 = blockIdx.y * 64  + (wave & 1) * 32;   // 2 waves along M
  const int n0 = blockIdx.x * 256 + (wave >> 1) * 64;  // 4 waves along N

  v8f acc[2][4] = {};

  const __bf16* ap0 = A + (size_t)(m0 + l16) * K + half * 16;
  const __bf16* ap1 = ap0 + 16 * K;
  const __bf16* wp  = W + (size_t)(n0 + l16) * K + half * 16;

  for (int kk = 0; kk < K; kk += 32) {
    // prefetch next K-tiles into cache (global_prefetch_b8)
    __builtin_prefetch(ap0 + kk + 256, 0, 3);
    __builtin_prefetch(ap1 + kk + 256, 0, 3);
    __builtin_prefetch(wp  + kk + 256, 0, 3);

    v16bf a[2], b[4];
    a[0] = load_a_frag(ap0 + kk);
    a[1] = load_a_frag(ap1 + kk);
#pragma unroll
    for (int ni = 0; ni < 4; ++ni)
      b[ni] = *(const v16bf*)(wp + (size_t)(ni * 16) * K + kk);
#pragma unroll
    for (int mi = 0; mi < 2; ++mi)
#pragma unroll
      for (int ni = 0; ni < 4; ++ni)
        acc[mi][ni] = __builtin_amdgcn_wmma_f32_16x16x32_bf16(
            false, a[mi], false, b[ni], (short)0, acc[mi][ni], false, false);
  }

  // C layout: VGPR e, half -> row = e + 8*half; lane%16 -> col.
#pragma unroll
  for (int mi = 0; mi < 2; ++mi)
#pragma unroll
    for (int ni = 0; ni < 4; ++ni) {
      float* op = out + (size_t)(m0 + mi * 16 + half * 8) * N + n0 + ni * 16 + l16;
#pragma unroll
      for (int e = 0; e < 8; ++e)
        op[(size_t)e * N] = acc[mi][ni][e];   // immediate offsets e*4096B
    }
}

// ---------------------------------------------------------------------------
// Leaky causal decay scan: k[t] = exp(-|beta|*10) * k[t-1] + kpre[t]
// kpre: (B,T,C) fp32 -> kscan: (B,NH,T,HS) fp32. One thread per (b,h,d).
// ---------------------------------------------------------------------------
__global__ __launch_bounds__(256)
void leaky_scan_kernel(const float* __restrict__ kpre, const float* __restrict__ beta_arr,
                       float* __restrict__ kscan) {
  int tid = blockIdx.x * 256 + threadIdx.x;       // 0 .. B*NH*HS-1
  int d = tid & (HS_ - 1);
  int h = (tid >> 6) & (NH_ - 1);
  int b = tid >> 10;
  float decay = expf(-fabsf(beta_arr[h]) * 10.0f);
  const float* src = kpre + (size_t)b * T_ * C_ + h * HS_ + d;
  float* dst = kscan + ((size_t)(b * NH_ + h) * T_) * HS_ + d;
  float acc = 0.0f;
  for (int t = 0; t < T_; ++t) {
    acc = acc * decay + src[(size_t)t * C_];
    dst[(size_t)t * HS_] = acc;
  }
}

// ---------------------------------------------------------------------------
// k normalize + scale. One wave per (b,h,t) row of 64. Writes fp32 k_cat and bf16 k.
// ---------------------------------------------------------------------------
__global__ __launch_bounds__(256)
void knorm_kernel(const float* __restrict__ ks, const float* __restrict__ key_scale,
                  float* __restrict__ kout, __bf16* __restrict__ kbf) {
  int gw   = (blockIdx.x * 256 + threadIdx.x) >> 5;   // (b*NH+h)*T + t
  int lane = threadIdx.x & 31;
  int h    = (gw / T_) % NH_;
  const float* row = ks + (size_t)gw * HS_;
  float e0 = row[lane], e1 = row[lane + 32];
  float s = e0 * e0 + e1 * e1;
#pragma unroll
  for (int m = 16; m >= 1; m >>= 1) s += __shfl_xor(s, m, 32);
  float sc = expf(fminf(10.0f * key_scale[h], KEY_SCALE_MAX)) / (sqrtf(s) + 1e-10f);
  float r0 = e0 * sc, r1 = e1 * sc;
  kout[(size_t)gw * HS_ + lane]      = r0;
  kout[(size_t)gw * HS_ + lane + 32] = r1;
  kbf [(size_t)gw * HS_ + lane]      = (__bf16)r0;
  kbf [(size_t)gw * HS_ + lane + 32] = (__bf16)r1;
}

// ---------------------------------------------------------------------------
// v mix (shift blend) + normalize + scale. One wave per (b,h,t).
// Writes fp32 v_cat (B,NH,T,HS) and bf16 V^T (B,NH,HS,T) for WMMA B-fragments.
// ---------------------------------------------------------------------------
__global__ __launch_bounds__(256)
void vprep_kernel(const float* __restrict__ vpre, const float* __restrict__ v_coef,
                  const float* __restrict__ val_scale,
                  float* __restrict__ vout, __bf16* __restrict__ vbft) {
  int gw   = (blockIdx.x * 256 + threadIdx.x) >> 5;
  int lane = threadIdx.x & 31;
  int t = gw % T_;
  int h = (gw / T_) % NH_;
  int b = gw / (T_ * NH_);
  float c = v_coef[h];
  const float* cur = vpre + ((size_t)b * T_ + t) * C_ + h * HS_;
  float c0 = cur[lane], c1 = cur[lane + 32];
  float n0 = 0.0f, n1 = 0.0f;
  if (t + 1 < T_) { n0 = cur[C_ + lane]; n1 = cur[C_ + lane + 32]; }
  float m0 = (1.0f - c) * n0 + c * c0;
  float m1 = (1.0f - c) * n1 + c * c1;
  float s = m0 * m0 + m1 * m1;
#pragma unroll
  for (int m = 16; m >= 1; m >>= 1) s += __shfl_xor(s, m, 32);
  float sc = expf(10.0f * val_scale[h]) / (sqrtf(s) + 1e-10f);
  m0 *= sc; m1 *= sc;
  vout[(size_t)gw * HS_ + lane]      = m0;
  vout[(size_t)gw * HS_ + lane + 32] = m1;
  size_t tb = ((size_t)(b * NH_ + h) * HS_) * T_;
  vbft[tb + (size_t)lane * T_ + t]        = (__bf16)m0;
  vbft[tb + (size_t)(lane + 32) * T_ + t] = (__bf16)m1;
}

// ---------------------------------------------------------------------------
// Flash attention, Q = K, mask: key >= query (attends to the future, per ref).
// One wave per 16-row query tile; key tiles of 32; online softmax.
// kbf: (B,NH,T,HS) bf16; vbft: (B,NH,HS,T) bf16; ybf out: (B,T,C) bf16.
// ---------------------------------------------------------------------------
__global__ __launch_bounds__(128)
void attn_kernel(const __bf16* __restrict__ kbf, const __bf16* __restrict__ vbft,
                 __bf16* __restrict__ ybf) {
  __shared__ float  sS[4][16 * 32];
  __shared__ __bf16 sP[4][16 * 32];

  const int w    = threadIdx.x >> 5;
  const int lane = threadIdx.x & 31;
  const int gw   = blockIdx.x * 4 + w;
  const int qi   = gw & (T_ / 16 - 1);
  const int h    = (gw >> 7) & (NH_ - 1);
  const int b    = gw >> 11;
  const int q0   = qi * 16;
  const int half = lane >> 4;
  const int l16  = lane & 15;

  const __bf16* Kb = kbf  + ((size_t)(b * NH_ + h) * T_) * HS_;
  const __bf16* Vb = vbft + ((size_t)(b * NH_ + h) * HS_) * T_;
  float*  S = sS[w];
  __bf16* P = sP[w];

  // Q fragments (A layout), HS=64 -> two K-steps of 32
  v16bf qf[2];
#pragma unroll
  for (int ds = 0; ds < 2; ++ds)
    qf[ds] = load_a_frag(Kb + (size_t)(q0 + l16) * HS_ + ds * 32 + half * 16);

  v8f yacc[4] = {};
  const int row = l16;               // lanes 16..31 duplicate rows 0..15 (benign)
  const int q   = q0 + row;
  float m_run = -3.0e38f, l_run = 0.0f;

  for (int j = (q0 & ~31); j < T_; j += 32) {
    // ---- S = Q @ K_tile^T : two 16x16 subtiles ----
#pragma unroll
    for (int kt = 0; kt < 2; ++kt) {
      v8f s = {};
#pragma unroll
      for (int ds = 0; ds < 2; ++ds) {
        v16bf kb = *(const v16bf*)(Kb + (size_t)(j + kt * 16 + l16) * HS_ +
                                   ds * 32 + half * 16);
        s = __builtin_amdgcn_wmma_f32_16x16x32_bf16(false, qf[ds], false, kb,
                                                    (short)0, s, false, false);
      }
#pragma unroll
      for (int e = 0; e < 8; ++e)
        S[(e + half * 8) * 32 + kt * 16 + l16] = s[e];
    }
    asm volatile("s_wait_dscnt 0" ::: "memory");

    // ---- online softmax over the 32 keys of this tile ----
    float rmax = -3.0e38f;
#pragma unroll 8
    for (int c = 0; c < 32; ++c) {
      float v = (j + c >= q) ? S[row * 32 + c] : -3.0e38f;
      rmax = fmaxf(rmax, v);
    }
    float m_new = fmaxf(m_run, rmax);
    float alpha = expf(m_run - m_new);
    float psum = 0.0f;
#pragma unroll 8
    for (int c = 0; c < 32; ++c) {
      float v = (j + c >= q) ? S[row * 32 + c] : -3.0e38f;
      float p = expf(v - m_new);
      psum += p;
      P[row * 32 + c] = (__bf16)p;
    }
    l_run = l_run * alpha + psum;
    m_run = m_new;
    asm volatile("s_wait_dscnt 0" ::: "memory");

    // ---- rescale accumulators by per-row alpha (broadcast from owning lane) ----
#pragma unroll
    for (int e = 0; e < 8; ++e) {
      float ae = __shfl(alpha, e + half * 8, 32);
#pragma unroll
      for (int n = 0; n < 4; ++n) yacc[n][e] *= ae;
    }

    // ---- y += P(16x32) @ V_tile(32x64) ----
    v16bf pf = load_a_frag(P + row * 32 + half * 16);
#pragma unroll
    for (int n = 0; n < 4; ++n) {
      v16bf vf = *(const v16bf*)(Vb + (size_t)(n * 16 + l16) * T_ + j + half * 16);
      yacc[n] = __builtin_amdgcn_wmma_f32_16x16x32_bf16(false, pf, false, vf,
                                                        (short)0, yacc[n], false, false);
    }
  }

  // ---- finalize: divide by l, write bf16 y in (B,T,C) layout ----
#pragma unroll
  for (int e = 0; e < 8; ++e) {
    float le  = __shfl(l_run, e + half * 8, 32);
    float inv = 1.0f / le;
    int rr = q0 + e + half * 8;
#pragma unroll
    for (int n = 0; n < 4; ++n)
      ybf[((size_t)b * T_ + rr) * C_ + h * HS_ + n * 16 + l16] =
          (__bf16)(yacc[n][e] * inv);
  }
}

// ---------------------------------------------------------------------------
extern "C" void kernel_launch(void* const* d_in, const int* in_sizes, int n_in,
                              void* d_out, int out_size, void* d_ws, size_t ws_size,
                              hipStream_t stream) {
  const float* x         = (const float*)d_in[0];
  const float* Wk        = (const float*)d_in[1];
  const float* Wv        = (const float*)d_in[2];
  const float* Wc        = (const float*)d_in[3];
  const float* beta      = (const float*)d_in[4];
  const float* key_scale = (const float*)d_in[5];
  const float* v_coef    = (const float*)d_in[6];
  const float* val_scale = (const float*)d_in[7];

  const size_t BTC = (size_t)B_ * T_ * C_;
  float* y_out = (float*)d_out;
  float* k_out = y_out + BTC;
  float* v_out = k_out + BTC;

  char* wsp = (char*)d_ws;
  size_t off = 0;
  auto walloc = [&](size_t bytes) -> void* {
    void* p = wsp + off;
    off = (off + bytes + 255) & ~(size_t)255;
    return p;
  };
  __bf16* xbf   = (__bf16*)walloc(BTC * 2);
  __bf16* wkbf  = (__bf16*)walloc((size_t)C_ * C_ * 2);
  __bf16* wvbf  = (__bf16*)walloc((size_t)C_ * C_ * 2);
  __bf16* wcbf  = (__bf16*)walloc((size_t)C_ * C_ * 2);
  float*  kpre  = (float*) walloc(BTC * 4);
  float*  vpre  = (float*) walloc(BTC * 4);
  float*  kscan = (float*) walloc(BTC * 4);
  __bf16* kbf   = (__bf16*)walloc(BTC * 2);
  __bf16* vbft  = (__bf16*)walloc(BTC * 2);
  __bf16* ybf   = (__bf16*)walloc(BTC * 2);

  const int nx = (int)BTC, nw = C_ * C_;
  f32_to_bf16_kernel<<<(nx + 255) / 256, 256, 0, stream>>>(x,  xbf,  nx);
  f32_to_bf16_kernel<<<(nw + 255) / 256, 256, 0, stream>>>(Wk, wkbf, nw);
  f32_to_bf16_kernel<<<(nw + 255) / 256, 256, 0, stream>>>(Wv, wvbf, nw);
  f32_to_bf16_kernel<<<(nw + 255) / 256, 256, 0, stream>>>(Wc, wcbf, nw);

  dim3 gg(C_ / 256, (B_ * T_) / 64);   // (4, 64)
  gemm_bf16_nt_kernel<<<gg, 256, 0, stream>>>(xbf, wkbf, kpre);
  gemm_bf16_nt_kernel<<<gg, 256, 0, stream>>>(xbf, wvbf, vpre);

  leaky_scan_kernel<<<(B_ * NH_ * HS_) / 256, 256, 0, stream>>>(kpre, beta, kscan);

  knorm_kernel<<<(B_ * NH_ * T_) / 8, 256, 0, stream>>>(kscan, key_scale, k_out, kbf);
  vprep_kernel<<<(B_ * NH_ * T_) / 8, 256, 0, stream>>>(vpre, v_coef, val_scale,
                                                        v_out, vbft);

  attn_kernel<<<(B_ * NH_ * (T_ / 16)) / 4, 128, 0, stream>>>(kbf, vbft, ybf);

  gemm_bf16_nt_kernel<<<gg, 256, 0, stream>>>(ybf, wcbf, y_out);
}
